// NanoMoEWhiteBox_62818191671491
// MI455X (gfx1250) — matrix operand order repaired and checked by reading, not compiled, not run on hardware
//
#include <hip/hip_runtime.h>
#include <math.h>

// ---------------- problem constants ----------------
#define NTOK   2048     // B*T tokens
#define DMODEL 128
#define TSEQ   256
#define BB     8
#define NH     4
#define HDIM   32       // head dim = D/H
#define NE     8
#define FF     512      // MoE hidden
#define NLAYER 4
#define VOC    50257
#define VOCP   50272    // padded to multiple of 16
#define NVT    (VOCP / 16)   // 3142 vocab tiles

typedef _Float16 h16;
typedef __attribute__((ext_vector_type(16))) _Float16 h16x16;
typedef __attribute__((ext_vector_type(8)))  _Float16 h16x8;
typedef __attribute__((ext_vector_type(8)))  float    f32x8;

union HU { h16x16 v; h16x8 h[2]; };

__device__ __forceinline__ f32x8 zero8() {
  f32x8 z;
#pragma unroll
  for (int i = 0; i < 8; ++i) z[i] = 0.0f;
  return z;
}

__device__ __forceinline__ f32x8 wmma16(h16x16 a, h16x16 b, f32x8 c) {
  // D = A(16x32 f16) * B(32x16 f16) + C(16x16 f32)
  return __builtin_amdgcn_wmma_f32_16x16x32_f16(false, a, false, b, (short)0, c,
                                                false, false);
}

// WMMA A operand: 16x32 f16 tile from row-major [.., ld] source.
// Lane layout (ISA 7.12.2): lanes 0-15 row=lane, K chunks {0..7,16..23};
// lanes 16-31 row=lane-16, K chunks {8..15,24..31}.
__device__ __forceinline__ h16x16 load_a(const h16* __restrict__ base, int row0,
                                         int ld, int k0) {
  int l = threadIdx.x & 31;
  int r = l & 15;
  int koff = (l < 16) ? 0 : 8;
  const h16* p = base + (size_t)(row0 + r) * ld + k0 + koff;
  HU u;
  u.h[0] = *(const h16x8*)(p);
  u.h[1] = *(const h16x8*)(p + 16);
  return u.v;
}

// WMMA B operand: 32x16 f16 tile from row-major [K, M] source.
// Lane l holds column col0+(l&15); K chunks {0..7,16..23} (lanes 0-15) or
// {8..15,24..31} (lanes 16-31), matching load_a.
__device__ __forceinline__ h16x16 load_b(const h16* __restrict__ base, int k0,
                                         int ld, int col0) {
  int l = threadIdx.x & 31;
  int koff = (l < 16) ? 0 : 8;
  const h16* p = base + (size_t)(k0 + koff) * ld + col0 + (l & 15);
  h16x16 v;
#pragma unroll
  for (int s = 0; s < 8; ++s) {
    v[s] = p[(size_t)s * ld];
    v[s + 8] = p[(size_t)(s + 16) * ld];
  }
  return v;
}

// ---------------- elementwise / prep kernels ----------------

__global__ void k_cvt(const float* __restrict__ src, h16* __restrict__ dst,
                      size_t n, size_t ntot) {
  size_t i = (size_t)blockIdx.x * blockDim.x + threadIdx.x;
  if (i >= ntot) return;
  dst[i] = (i < n) ? (h16)src[i] : (h16)0.0f;
}

// x[d, n] = tok_emb[idx[n], d] + pos_emb[t, d]   (feature-major f32)
__global__ void k_embed(const int* __restrict__ idx,
                        const float* __restrict__ tok,
                        const float* __restrict__ pos,
                        float* __restrict__ x) {
  int i = blockIdx.x * blockDim.x + threadIdx.x;
  int n = i & (NTOK - 1);
  int d = i >> 11;           // / NTOK
  if (d >= DMODEL) return;
  int t = n & (TSEQ - 1);
  x[(size_t)d * NTOK + n] = tok[(size_t)idx[n] * DMODEL + d] + pos[t * DMODEL + d];
}

// LayerNorm over features for each token; feature-major in/out, f16 out.
__global__ void k_ln(const float* __restrict__ x, const float* __restrict__ g,
                     const float* __restrict__ b, h16* __restrict__ out) {
  int n = blockIdx.x * blockDim.x + threadIdx.x;
  if (n >= NTOK) return;
  float s = 0.f, s2 = 0.f;
  for (int d = 0; d < DMODEL; ++d) {
    float v = x[(size_t)d * NTOK + n];
    s += v; s2 += v * v;
  }
  float mu = s * (1.0f / DMODEL);
  float var = s2 * (1.0f / DMODEL) - mu * mu;
  float rstd = rsqrtf(var + 1e-5f);
  for (int d = 0; d < DMODEL; ++d) {
    float v = (x[(size_t)d * NTOK + n] - mu) * rstd * g[d] + b[d];
    out[(size_t)d * NTOK + n] = (h16)v;
  }
}

// ---------------- GEMM kernels (wave = 16x16 tiles, computes C^T) ----------

// QKV: [N=384] = in_w[l] (384x128) x xl^T.  Epilogue scatters into
// qT [B,H,32,T], kR [B,H,T,32], vT [B,H,32,T] (f16).
__global__ void k_qkv(const h16* __restrict__ Wl, const float* __restrict__ biasl,
                      const h16* __restrict__ xl, h16* __restrict__ qT,
                      h16* __restrict__ kR, h16* __restrict__ vT) {
  int w = threadIdx.x >> 5, lane = threadIdx.x & 31;
  int ft = blockIdx.y * 8 + w;   // 0..23 feature tiles
  int mt = blockIdx.x;           // 0..127 token tiles
  f32x8 acc = zero8();
#pragma unroll
  for (int kk = 0; kk < 4; ++kk)
    acc = wmma16(load_a(Wl, ft * 16, DMODEL, kk * 32),
                 load_b(xl, kk * 32, NTOK, mt * 16), acc);
  int qoff = lane & 15, g2 = lane >> 4;
  int n = mt * 16 + qoff, btok = n / TSEQ, t = n & (TSEQ - 1);
  int fbase = ft * 16 + g2 * 8;
#pragma unroll
  for (int r = 0; r < 8; ++r) {
    int f = fbase + r;
    float v = acc[r] + biasl[f];
    int fm = f & 127, h = fm >> 5, dd = fm & 31;
    size_t bh = (size_t)(btok * NH + h);
    if (f < 128)       qT[(bh * HDIM + dd) * TSEQ + t] = (h16)v;
    else if (f < 256)  kR[(bh * TSEQ + t) * HDIM + dd] = (h16)v;
    else               vT[(bh * HDIM + dd) * TSEQ + t] = (h16)v;
  }
}

// Fused attention for one (b,h): S^T = K x q^T via WMMA (K-dim = 32),
// softmax in regs, P^T f16 staged in LDS, O^T = V^T x P^T via WMMA.
__global__ void k_attn(const h16* __restrict__ qT, const h16* __restrict__ kR,
                       const h16* __restrict__ vT, h16* __restrict__ oT) {
  __shared__ h16 Plds[4][TSEQ * 16];   // per-wave 256x16 f16 tile (32 KB)
  int w = threadIdx.x >> 5, lane = threadIdx.x & 31;
  int bh = blockIdx.x, btok = bh / NH, h = bh % NH;
  const h16* q = qT + (size_t)bh * HDIM * TSEQ;
  const h16* k = kR + (size_t)bh * TSEQ * HDIM;
  const h16* v = vT + (size_t)bh * HDIM * TSEQ;
  int qoff = lane & 15, g2 = lane >> 4;
  const float scale = 0.17677669529663687f;  // 1/sqrt(32)
  for (int qt = w; qt < 16; qt += 4) {
    h16x16 qb = load_b(q, 0, TSEQ, qt * 16);     // dh x 16 queries
    f32x8 s[16];
#pragma unroll
    for (int j = 0; j < 16; ++j)
      s[j] = wmma16(load_a(k, j * 16, HDIM, 0), qb, zero8());
    // softmax over keys (query = lane&15, halves combined via shfl_xor 16)
    float mx = -3.4e38f;
#pragma unroll
    for (int j = 0; j < 16; ++j)
#pragma unroll
      for (int r = 0; r < 8; ++r) mx = fmaxf(mx, s[j][r]);
    mx = fmaxf(mx, __shfl_xor(mx, 16, 32));
    float sum = 0.f;
#pragma unroll
    for (int j = 0; j < 16; ++j)
#pragma unroll
      for (int r = 0; r < 8; ++r) {
        float p = __expf(scale * (s[j][r] - mx));
        s[j][r] = p;
        sum += p;
      }
    sum += __shfl_xor(sum, 16, 32);
    float rs = 1.0f / sum;
    h16* pl = Plds[w];
#pragma unroll
    for (int j = 0; j < 16; ++j)
#pragma unroll
      for (int r = 0; r < 8; ++r)
        pl[(16 * j + 8 * g2 + r) * 16 + qoff] = (h16)s[j][r];
    // O^T = V^T (dh x T) * P^T (T x 16)
#pragma unroll
    for (int i = 0; i < 2; ++i) {
      f32x8 o = zero8();
#pragma unroll
      for (int c = 0; c < 8; ++c)
        o = wmma16(load_a(v, i * 16, TSEQ, c * 32),
                   load_b(pl, c * 32, 16, 0), o);
#pragma unroll
      for (int r = 0; r < 8; ++r) {
        int dd = i * 16 + 8 * g2 + r;
        oT[(size_t)(h * HDIM + dd) * NTOK + btok * TSEQ + qt * 16 + qoff] =
            (h16)(o[r] * rs);
      }
    }
  }
}

// Output projection with residual accumulate into x (feature-major f32).
__global__ void k_outproj(const h16* __restrict__ Wl,
                          const float* __restrict__ biasl,
                          const h16* __restrict__ oT, float* __restrict__ x) {
  int w = threadIdx.x >> 5, lane = threadIdx.x & 31;
  int dt = w, mt = blockIdx.x;
  f32x8 acc = zero8();
#pragma unroll
  for (int kk = 0; kk < 4; ++kk)
    acc = wmma16(load_a(Wl, dt * 16, DMODEL, kk * 32),
                 load_b(oT, kk * 32, NTOK, mt * 16), acc);
  int qoff = lane & 15, g2 = lane >> 4;
  int n = mt * 16 + qoff, dbase = dt * 16 + g2 * 8;
#pragma unroll
  for (int r = 0; r < 8; ++r) {
    int d = dbase + r;
    x[(size_t)d * NTOK + n] += acc[r] + biasl[d];
  }
}

// Gate: per-token softmax over E logits, top-2 renormalized -> we [E, NTOK].
__global__ void k_gate(const h16* __restrict__ x2, const float* __restrict__ gw,
                       float* __restrict__ we) {
  int n = blockIdx.x * blockDim.x + threadIdx.x;
  if (n >= NTOK) return;
  float p[NE];
  float mx = -3.4e38f;
  for (int e = 0; e < NE; ++e) {
    float s = 0.f;
    for (int d = 0; d < DMODEL; ++d)
      s += (float)x2[(size_t)d * NTOK + n] * gw[e * DMODEL + d];
    p[e] = s;
    mx = fmaxf(mx, s);
  }
  float sum = 0.f;
  for (int e = 0; e < NE; ++e) { p[e] = __expf(p[e] - mx); sum += p[e]; }
  for (int e = 0; e < NE; ++e) p[e] /= sum;
  int i0 = 0;
  for (int e = 1; e < NE; ++e) if (p[e] > p[i0]) i0 = e;
  int i1 = (i0 == 0) ? 1 : 0;
  for (int e = 0; e < NE; ++e) if (e != i0 && p[e] > p[i1]) i1 = e;
  float s2 = p[i0] + p[i1];
  for (int e = 0; e < NE; ++e) we[(size_t)e * NTOK + n] = 0.f;
  we[(size_t)i0 * NTOK + n] = p[i0] / s2;
  we[(size_t)i1 * NTOK + n] = p[i1] / s2;
}

// MoE first GEMM + SiLU: hT[e, :, n] = silu(w1[e] x x2^T), per-expert grid.z.
__global__ void k_moe1(const h16* __restrict__ w1l, const h16* __restrict__ x2,
                       h16* __restrict__ hT) {
  int w = threadIdx.x >> 5, lane = threadIdx.x & 31;
  int mt = blockIdx.x, ht = blockIdx.y * 8 + w, e = blockIdx.z;
  const h16* We = w1l + (size_t)e * FF * DMODEL;
  f32x8 acc = zero8();
#pragma unroll
  for (int kk = 0; kk < 4; ++kk)
    acc = wmma16(load_a(We, ht * 16, DMODEL, kk * 32),
                 load_b(x2, kk * 32, NTOK, mt * 16), acc);
  int qoff = lane & 15, g2 = lane >> 4;
  int n = mt * 16 + qoff;
#pragma unroll
  for (int r = 0; r < 8; ++r) {
    int hr = ht * 16 + g2 * 8 + r;
    float v = acc[r];
    v = v / (1.0f + __expf(-v));  // silu
    hT[((size_t)e * FF + hr) * NTOK + n] = (h16)v;
  }
}

// MoE second GEMM: x += sum_e we[e,n] * (w2[e] x h_e^T).
// The 8 waves of a block share one token tile, so the h_e^T slab (512x16 f16,
// 16 KB) is staged in LDS once per block instead of 8 redundant global reads.
__global__ void k_moe2(const h16* __restrict__ w2l, const h16* __restrict__ hT,
                       const float* __restrict__ we, float* __restrict__ x) {
  __shared__ h16 Bt[FF][16];   // 16 KB staged B operand
  int w = threadIdx.x >> 5, lane = threadIdx.x & 31;
  int dt = w, mt = blockIdx.x;
  int qoff = lane & 15, g2 = lane >> 4;
  int n = mt * 16 + qoff;
  f32x8 tot = zero8();
  for (int e = 0; e < NE; ++e) {
    __syncthreads();   // previous iteration's readers are done
    const h16* He = hT + (size_t)e * FF * NTOK + (size_t)mt * 16;
    for (int r = threadIdx.x; r < FF; r += 256)
      *(h16x16*)(&Bt[r][0]) = *(const h16x16*)(He + (size_t)r * NTOK);
    __syncthreads();
    f32x8 acc = zero8();
    const h16* We = w2l + (size_t)e * DMODEL * FF;
#pragma unroll
    for (int kk = 0; kk < 16; ++kk)
      acc = wmma16(load_a(We, dt * 16, FF, kk * 32),
                   load_b(&Bt[0][0], kk * 32, 16, 0), acc);
    float wg = we[(size_t)e * NTOK + n];
#pragma unroll
    for (int r = 0; r < 8; ++r) tot[r] += wg * acc[r];
  }
#pragma unroll
  for (int r = 0; r < 8; ++r) {
    int d = dt * 16 + g2 * 8 + r;
    x[(size_t)d * NTOK + n] += tot[r];
  }
}

// LM head: out[n, v] = xf^T x lm_w^T.  Each wave keeps the 4 activation (B)
// fragments in registers and sweeps 4 vocab tiles (4x fewer B loads, streams
// lm_w once); LDS-transposed epilogue gives contiguous f32 stores along vocab
// (the 412 MB HBM store stream is the roofline floor).
__global__ void k_lmhead(const h16* __restrict__ lmw, const h16* __restrict__ xf,
                         float* __restrict__ out) {
  __shared__ float tile[8][16][20];   // [wave][token][voc], padded
  int w = threadIdx.x >> 5, lane = threadIdx.x & 31;
  int mt = blockIdx.x;
  int vt0 = (blockIdx.y * 8 + w) * 4;
  int qoff = lane & 15, g2 = lane >> 4;
  h16x16 b0 = load_b(xf, 0,  NTOK, mt * 16);
  h16x16 b1 = load_b(xf, 32, NTOK, mt * 16);
  h16x16 b2 = load_b(xf, 64, NTOK, mt * 16);
  h16x16 b3 = load_b(xf, 96, NTOK, mt * 16);
  int tt = lane & 15, half = lane >> 4;
  int token = mt * 16 + tt;
  for (int u = 0; u < 4; ++u) {
    int vt = vt0 + u;
    if (vt >= NVT) break;
    if (vt + 1 < NVT)  // prefetch next weight tile into cache hierarchy
      __builtin_prefetch(lmw + (size_t)(vt + 1) * 16 * DMODEL, 0, 1);
    f32x8 acc = zero8();
    acc = wmma16(load_a(lmw, vt * 16, DMODEL, 0),  b0, acc);
    acc = wmma16(load_a(lmw, vt * 16, DMODEL, 32), b1, acc);
    acc = wmma16(load_a(lmw, vt * 16, DMODEL, 64), b2, acc);
    acc = wmma16(load_a(lmw, vt * 16, DMODEL, 96), b3, acc);
#pragma unroll
    for (int r = 0; r < 8; ++r) tile[w][qoff][g2 * 8 + r] = acc[r];
    // same-wave LDS transpose (DS ops are in-order within a wave)
    int vbase = vt * 16 + half * 8;
#pragma unroll
    for (int j = 0; j < 8; ++j) {
      int voc = vbase + j;
      if (voc < VOC) out[(size_t)token * VOC + voc] = tile[w][tt][half * 8 + j];
    }
  }
}

// ---------------- host launcher ----------------

extern "C" void kernel_launch(void* const* d_in, const int* in_sizes, int n_in,
                              void* d_out, int out_size, void* d_ws, size_t ws_size,
                              hipStream_t stream) {
  const int*   idx     = (const int*)  d_in[0];
  const float* tok_emb = (const float*)d_in[1];
  const float* pos_emb = (const float*)d_in[2];
  const float* ln1_g   = (const float*)d_in[3];
  const float* ln1_b   = (const float*)d_in[4];
  const float* in_w    = (const float*)d_in[5];
  const float* in_b    = (const float*)d_in[6];
  const float* out_w   = (const float*)d_in[7];
  const float* out_b   = (const float*)d_in[8];
  const float* ln2_g   = (const float*)d_in[9];
  const float* ln2_b   = (const float*)d_in[10];
  const float* gate_w  = (const float*)d_in[11];
  const float* w1      = (const float*)d_in[12];
  const float* w2      = (const float*)d_in[13];
  const float* lnf_g   = (const float*)d_in[14];
  const float* lnf_b   = (const float*)d_in[15];
  const float* lm_w    = (const float*)d_in[16];
  float* out = (float*)d_out;

  char* base = (char*)d_ws;
  size_t off = 0;
  auto carve = [&](size_t bytes) -> void* {
    void* r = base + off;
    off += (bytes + 255) & ~(size_t)255;
    return r;
  };
  const size_t INW = (size_t)NLAYER * 3 * DMODEL * DMODEL;   // 196608
  const size_t OUW = (size_t)NLAYER * DMODEL * DMODEL;       // 65536
  const size_t W1N = (size_t)NLAYER * NE * FF * DMODEL;      // 2097152
  const size_t W2N = (size_t)NLAYER * NE * DMODEL * FF;      // 2097152
  const size_t LMN = (size_t)VOC * DMODEL;
  const size_t LMP = (size_t)VOCP * DMODEL;

  h16*   inw_h  = (h16*)carve(INW * 2);
  h16*   outw_h = (h16*)carve(OUW * 2);
  h16*   w1_h   = (h16*)carve(W1N * 2);
  h16*   w2_h   = (h16*)carve(W2N * 2);
  h16*   lm_h   = (h16*)carve(LMP * 2);
  float* x      = (float*)carve((size_t)DMODEL * NTOK * 4);
  h16*   xl     = (h16*)carve((size_t)DMODEL * NTOK * 2);
  h16*   x2     = (h16*)carve((size_t)DMODEL * NTOK * 2);
  h16*   xf     = (h16*)carve((size_t)DMODEL * NTOK * 2);
  h16*   oT     = (h16*)carve((size_t)DMODEL * NTOK * 2);
  h16*   qT     = (h16*)carve((size_t)BB * NH * HDIM * TSEQ * 2);
  h16*   kR     = (h16*)carve((size_t)BB * NH * TSEQ * HDIM * 2);
  h16*   vT     = (h16*)carve((size_t)BB * NH * HDIM * TSEQ * 2);
  float* wegate = (float*)carve((size_t)NE * NTOK * 4);
  h16*   hT     = (h16*)carve((size_t)NE * FF * NTOK * 2);

  auto cvt = [&](const float* s, h16* dp, size_t n, size_t ntot) {
    int blocks = (int)((ntot + 255) / 256);
    k_cvt<<<blocks, 256, 0, stream>>>(s, dp, n, ntot);
  };
  cvt(in_w, inw_h, INW, INW);
  cvt(out_w, outw_h, OUW, OUW);
  cvt(w1, w1_h, W1N, W1N);
  cvt(w2, w2_h, W2N, W2N);
  cvt(lm_w, lm_h, LMN, LMP);   // zero-pad vocab rows 50257..50271

  k_embed<<<(DMODEL * NTOK) / 256, 256, 0, stream>>>(idx, tok_emb, pos_emb, x);

  for (int l = 0; l < NLAYER; ++l) {
    const h16* inw_l  = inw_h  + (size_t)l * 3 * DMODEL * DMODEL;
    const h16* outw_l = outw_h + (size_t)l * DMODEL * DMODEL;
    const h16* w1_l   = w1_h   + (size_t)l * NE * FF * DMODEL;
    const h16* w2_l   = w2_h   + (size_t)l * NE * DMODEL * FF;

    k_ln<<<NTOK / 256, 256, 0, stream>>>(x, ln1_g + l * DMODEL,
                                         ln1_b + l * DMODEL, xl);
    k_qkv<<<dim3(NTOK / 16, 3), 256, 0, stream>>>(inw_l, in_b + l * 3 * DMODEL,
                                                  xl, qT, kR, vT);
    k_attn<<<BB * NH, 128, 0, stream>>>(qT, kR, vT, oT);
    k_outproj<<<NTOK / 16, 256, 0, stream>>>(outw_l, out_b + l * DMODEL, oT, x);
    k_ln<<<NTOK / 256, 256, 0, stream>>>(x, ln2_g + l * DMODEL,
                                         ln2_b + l * DMODEL, x2);
    k_gate<<<NTOK / 256, 256, 0, stream>>>(x2, gate_w + (size_t)l * NE * DMODEL,
                                           wegate);
    k_moe1<<<dim3(NTOK / 16, 4, NE), 256, 0, stream>>>(w1_l, x2, hT);
    k_moe2<<<NTOK / 16, 256, 0, stream>>>(w2_l, hT, wegate, x);
  }

  k_ln<<<NTOK / 256, 256, 0, stream>>>(x, lnf_g, lnf_b, xf);
  k_lmhead<<<dim3(NTOK / 16, (NVT + 31) / 32), 256, 0, stream>>>(lm_h, xf, out);
}
